// ArithmeticUserStateModel_17411797418062
// MI455X (gfx1250) — matrix operand order repaired and checked by reading, not compiled, not run on hardware
//
#include <hip/hip_runtime.h>

// ---------------------------------------------------------------------------
// Problem constants (reference: B=4096, T=256, PD=32, SD=64, NB=41)
// ---------------------------------------------------------------------------
#define BB 4096
#define TT 256
#define PD 32
#define SD 64
#define NB 41

typedef __bf16 bf16;
typedef __attribute__((ext_vector_type(16))) __bf16 v16bf;
typedef __attribute__((ext_vector_type(8)))  __bf16 v8bf;
typedef __attribute__((ext_vector_type(8)))  float  v8f;

// Fragment table layout in d_ws (each fragment = 512 bf16 = 1024 B):
//   B-fragment of a (32 x 16) K-by-N weight tile, element e of lane L maps to
//   K = 8*(L/16) + e%8 + 16*(e/8), N = L%16  (matches v_wmma 16-bit B operand layout).
#define F_ENC1 0     // enc_w1  (K=3 ->pad32, N=32)      : 2 frags
#define F_ENC2 2     // enc_w2  (K=32, N=32)             : 2 frags
#define F_WC   4     // proj_w@gru_wih (K=34->64, N=192) : 12 tiles x 2 k-chunks = 24
#define F_WHH  28    // gru_whh (K=64, N=192)            : 24
#define F_PW1  52    // pred_w1 (K=96, N=64)             : 4 x 3 = 12
#define F_PW2  64    // pred_w2 (K=64, N=64)             : 4 x 2 = 8
#define F_ANS  72    // ans_w   (K=64, N=41->48)         : 3 x 2 = 6
#define F_COR  78    // cor_w   (K=64, N=1->16)          : 2
#define NFRAGS 80
#define FRAG_BYTES (NFRAGS * 512 * 2)

// Bias block (512 floats) appended after the fragments in d_ws:
//   [0,32)   enc_b1          [32,64)  enc_b2
//   [64,128) bcr = projb@wih + bih + bhh  (r gate)
//   [128,192)bcz = projb@wih + bih + bhh  (z gate)
//   [192,256)bxn = projb@wih + bih        (n gate, x part)
//   [256,320)bhn = bhh                    (n gate, h part, multiplied by r)
//   [320,384)pred_b1         [384,448) pred_b2
//   [448,496)ans_b (pad48)   [496,512) cor_b (pad16)
#define BIAS_FLOATS 512

__device__ inline v8f wmma_bf16(v16bf a, v16bf b, v8f c) {
  return __builtin_amdgcn_wmma_f32_16x16x32_bf16(
      /*neg_a=*/false, a, /*neg_b=*/false, b,
      /*c_mod=*/(short)0, c, /*reuse_a=*/false, /*reuse_b=*/false);
}

// A fragment (16 x 32 bf16, M x K) from row-major [16][ld] bf16 LDS storage.
__device__ inline v16bf load_afrag(const bf16* base, int ld, int k0, int lane) {
  int m  = lane & 15;
  int kb = (lane >> 4) * 8;
  const bf16* p = base + m * ld + k0 + kb;
  v8bf lo = *(const v8bf*)(p);        // K = kb .. kb+7
  v8bf hi = *(const v8bf*)(p + 16);   // K = kb+16 .. kb+23
  return __builtin_shufflevector(lo, hi, 0, 1, 2, 3, 4, 5, 6, 7,
                                         8, 9, 10, 11, 12, 13, 14, 15);
}

// B fragment: pre-swizzled, lane-contiguous (32 B per lane), served from LDS.
__device__ inline v16bf load_bfrag(const bf16* frag, int lane) {
  const bf16* p = frag + lane * 16;
  v8bf lo = *(const v8bf*)(p);
  v8bf hi = *(const v8bf*)(p + 8);
  return __builtin_shufflevector(lo, hi, 0, 1, 2, 3, 4, 5, 6, 7,
                                         8, 9, 10, 11, 12, 13, 14, 15);
}

__device__ inline float sigmoidf_(float x) { return 1.0f / (1.0f + expf(-x)); }

// ---------------------------------------------------------------------------
// Prep kernel: fold proj/gru biases, build Wc = proj_w @ gru_wih, and swizzle
// every weight matrix into bf16 WMMA B-fragments.
// ---------------------------------------------------------------------------
__global__ void prep_weights(const float* enc_w1, const float* enc_w2,
                             const float* proj_w, const float* proj_b,
                             const float* wih, const float* bih,
                             const float* whh, const float* bhh,
                             const float* pw1, const float* pw2,
                             const float* answ, const float* corw,
                             const float* enc_b1, const float* enc_b2,
                             const float* pb1, const float* pb2,
                             const float* ansb, const float* corb,
                             bf16* frags, float* biases) {
  int tid = threadIdx.x;
  for (int idx = tid; idx < NFRAGS * 512; idx += blockDim.x) {
    int f = idx >> 9;
    int e = idx & 511;
    int L = e >> 4, el = e & 15;
    int krel = ((L >> 4) << 3) + (el & 7) + ((el >> 3) << 4);  // 8*(L/16) + e%8 + 16*(e/8)
    int nrel = L & 15;
    float val = 0.0f;
    if (f < F_ENC2) {                       // enc_w1: K=3, N=32
      int k = krel, n = (f - F_ENC1) * 16 + nrel;
      if (k < 3) val = enc_w1[k * 32 + n];
    } else if (f < F_WC) {                  // enc_w2: K=32, N=32
      int k = krel, n = (f - F_ENC2) * 16 + nrel;
      val = enc_w2[k * 32 + n];
    } else if (f < F_WHH) {                 // Wc = proj_w @ wih: K=34, N=192
      int g = f - F_WC, j = g >> 1, kc = g & 1;
      int k = kc * 32 + krel, n = j * 16 + nrel;
      if (k < PD + 2) {
        float s = 0.0f;
        for (int q = 0; q < SD; ++q) s += proj_w[k * SD + q] * wih[q * 192 + n];
        val = s;
      }
    } else if (f < F_PW1) {                 // whh: K=64, N=192
      int g = f - F_WHH, j = g >> 1, kc = g & 1;
      val = whh[(kc * 32 + krel) * 192 + j * 16 + nrel];
    } else if (f < F_PW2) {                 // pred_w1: K=96, N=64
      int g = f - F_PW1, nt = g / 3, kc = g % 3;
      val = pw1[(kc * 32 + krel) * SD + nt * 16 + nrel];
    } else if (f < F_ANS) {                 // pred_w2: K=64, N=64
      int g = f - F_PW2, nt = g >> 1, kc = g & 1;
      val = pw2[(kc * 32 + krel) * SD + nt * 16 + nrel];
    } else if (f < F_COR) {                 // ans_w: K=64, N=41
      int g = f - F_ANS, nt = g >> 1, kc = g & 1;
      int k = kc * 32 + krel, n = nt * 16 + nrel;
      if (n < NB) val = answ[k * NB + n];
    } else {                                // cor_w: K=64, N=1
      int kc = (f - F_COR) & 1;
      int k = kc * 32 + krel, n = nrel;
      if (n < 1) val = corw[k];
    }
    frags[idx] = (bf16)val;
  }
  for (int i = tid; i < BIAS_FLOATS; i += blockDim.x) {
    float v = 0.0f;
    if (i < 32) v = enc_b1[i];
    else if (i < 64) v = enc_b2[i - 32];
    else if (i < 320) {
      int n;
      if (i < 128)      n = i - 64;          // r gate col
      else if (i < 192) n = i - 128 + 64;    // z gate col
      else if (i < 256) n = i - 192 + 128;   // n gate (x part) col
      else { v = bhh[128 + (i - 256)]; biases[i] = v; continue; }
      float s = 0.0f;
      for (int q = 0; q < SD; ++q) s += proj_b[q] * wih[q * 192 + n];
      v = s + bih[n];
      if (i < 192) v += bhh[n];              // fold bhh for r/z only
    } else if (i < 384) v = pb1[i - 320];
    else if (i < 448) v = pb2[i - 384];
    else if (i < 496) { int n = i - 448; v = (n < NB) ? ansb[n] : 0.0f; }
    else { v = (i == 496) ? corb[0] : 0.0f; }
    biases[i] = v;
  }
}

// ---------------------------------------------------------------------------
// Fused scan kernel: one 128-thread block (4 wave32) per 16 rows of B.
// All weight fragments are staged once into (dynamic) LDS; the 256-step
// serial loop then touches global memory only for inputs and logit stores.
// ---------------------------------------------------------------------------
extern __shared__ __align__(16) bf16 sfrag[];   // NFRAGS * 512 bf16 (80 KB)

__global__ __launch_bounds__(128) void
fused_scan(const float* __restrict__ problems, const float* __restrict__ answers,
           const float* __restrict__ corrects, const bf16* __restrict__ frags,
           const float* __restrict__ gbias, float* __restrict__ out_ans,
           float* __restrict__ out_cor) {
  __shared__ __align__(16) bf16 Acat[16 * 96];  // cols 0-63: h(bf16), 64-95: pe
  __shared__ __align__(16) bf16 Agx[16 * 64];   // 0-31 pe, 32 ans, 33 cor, rest 0
  __shared__ __align__(16) bf16 Aenc[16 * 32];  // cols 0-2 problem, rest 0
  __shared__ __align__(16) bf16 Ah1[16 * 32];
  __shared__ __align__(16) bf16 F1[16 * 64];
  __shared__ __align__(16) bf16 F2[16 * 64];
  __shared__ float hf[16 * 64];                 // f32 master copy of h
  __shared__ float sb[BIAS_FLOATS];

  const int tid  = threadIdx.x;
  const int lane = tid & 31;
  const int wid  = tid >> 5;       // wave id 0..3
  const int row0 = blockIdx.x * 16;
  const int ncol = lane & 15;      // C/D fragment column within tile
  const int mrow0 = (lane >> 4) * 8;

  // one-time: stage all weight fragments into LDS (16 B per thread per iter)
  {
    v8bf* dst = (v8bf*)sfrag;
    const v8bf* src = (const v8bf*)frags;
    for (int i = tid; i < NFRAGS * 512 / 8; i += 128) dst[i] = src[i];
  }
  for (int i = tid; i < 16 * 96; i += 128) Acat[i] = (bf16)0.0f;
  for (int i = tid; i < 16 * 64; i += 128) { Agx[i] = (bf16)0.0f; hf[i] = 0.0f; }
  for (int i = tid; i < 16 * 32; i += 128) Aenc[i] = (bf16)0.0f;
  for (int i = tid; i < BIAS_FLOATS; i += 128) sb[i] = gbias[i];
  __syncthreads();

  for (int t = 0; t < TT; ++t) {
    // ---- stage in this step's observations (wave 0, lanes 0-15) ----
    if (wid == 0 && lane < 16) {
      int m = lane;
      long base = (long)(row0 + m) * TT + t;
      const float* pp = problems + base * 3;
      Aenc[m * 32 + 0] = (bf16)pp[0];
      Aenc[m * 32 + 1] = (bf16)pp[1];
      Aenc[m * 32 + 2] = (bf16)pp[2];
      Agx[m * 64 + 32] = (bf16)answers[base];
      Agx[m * 64 + 33] = (bf16)corrects[base];
    }
    __syncthreads();

    // ---- encoder layer 1: waves 0,1 ----
    if (wid < 2) {
      v8f c = {};
      c = wmma_bf16(load_afrag(Aenc, 32, 0, lane),
                    load_bfrag(sfrag + (F_ENC1 + wid) * 512, lane), c);
      int col = wid * 16 + ncol;
      float b = sb[col];
#pragma unroll
      for (int v = 0; v < 8; ++v)
        Ah1[(v + mrow0) * 32 + col] = (bf16)fmaxf(c[v] + b, 0.0f);
    }
    __syncthreads();

    // ---- encoder layer 2: waves 2,3 -> pe into Agx and Acat ----
    if (wid >= 2) {
      int nt = wid - 2;
      v8f c = {};
      c = wmma_bf16(load_afrag(Ah1, 32, 0, lane),
                    load_bfrag(sfrag + (F_ENC2 + nt) * 512, lane), c);
      int col = nt * 16 + ncol;
      float b = sb[32 + col];
#pragma unroll
      for (int v = 0; v < 8; ++v) {
        bf16 q = (bf16)fmaxf(c[v] + b, 0.0f);
        int m = v + mrow0;
        Agx[m * 64 + col]       = q;
        Acat[m * 96 + 64 + col] = q;
      }
    }
    __syncthreads();

    // ---- pred layer 1 ([h|pe] @ pred_w1, K=96) + gx (Agx @ Wc) ----
    v8f f1 = {};
#pragma unroll
    for (int kc = 0; kc < 3; ++kc)
      f1 = wmma_bf16(load_afrag(Acat, 96, 32 * kc, lane),
                     load_bfrag(sfrag + (F_PW1 + wid * 3 + kc) * 512, lane), f1);
    {
      int col = wid * 16 + ncol;
      float b = sb[320 + col];
#pragma unroll
      for (int v = 0; v < 8; ++v)
        F1[(v + mrow0) * 64 + col] = (bf16)fmaxf(f1[v] + b, 0.0f);
    }
    v8f gx[3] = {{}, {}, {}};
#pragma unroll
    for (int g = 0; g < 3; ++g)
#pragma unroll
      for (int kc = 0; kc < 2; ++kc)
        gx[g] = wmma_bf16(load_afrag(Agx, 64, 32 * kc, lane),
                          load_bfrag(sfrag + (F_WC + (4 * g + wid) * 2 + kc) * 512, lane),
                          gx[g]);
    __syncthreads();

    // ---- pred layer 2 + gh = h @ whh (h still untouched) ----
    v8f f2 = {};
#pragma unroll
    for (int kc = 0; kc < 2; ++kc)
      f2 = wmma_bf16(load_afrag(F1, 64, 32 * kc, lane),
                     load_bfrag(sfrag + (F_PW2 + wid * 2 + kc) * 512, lane), f2);
    {
      int col = wid * 16 + ncol;
      float b = sb[384 + col];
#pragma unroll
      for (int v = 0; v < 8; ++v)
        F2[(v + mrow0) * 64 + col] = (bf16)fmaxf(f2[v] + b, 0.0f);
    }
    v8f gh[3] = {{}, {}, {}};
#pragma unroll
    for (int g = 0; g < 3; ++g)
#pragma unroll
      for (int kc = 0; kc < 2; ++kc)
        gh[g] = wmma_bf16(load_afrag(Acat, 96, 32 * kc, lane),
                          load_bfrag(sfrag + (F_WHH + (4 * g + wid) * 2 + kc) * 512, lane),
                          gh[g]);
    __syncthreads();

    // ---- logits: waves 0-2 answer tiles, wave 3 correctness ----
    if (wid < 3) {
      v8f oa = {};
#pragma unroll
      for (int kc = 0; kc < 2; ++kc)
        oa = wmma_bf16(load_afrag(F2, 64, 32 * kc, lane),
                       load_bfrag(sfrag + (F_ANS + wid * 2 + kc) * 512, lane), oa);
      int col = wid * 16 + ncol;
      if (col < NB) {
        float b = sb[448 + col];
#pragma unroll
        for (int v = 0; v < 8; ++v) {
          long row = row0 + v + mrow0;
          out_ans[(row * TT + t) * NB + col] = oa[v] + b;
        }
      }
    } else {
      v8f oc = {};
#pragma unroll
      for (int kc = 0; kc < 2; ++kc)
        oc = wmma_bf16(load_afrag(F2, 64, 32 * kc, lane),
                       load_bfrag(sfrag + (F_COR + kc) * 512, lane), oc);
      if (ncol == 0) {
        float b = sb[496];
#pragma unroll
        for (int v = 0; v < 8; ++v) {
          long row = row0 + v + mrow0;
          out_cor[row * TT + t] = oc[v] + b;
        }
      }
    }

    // ---- GRU gate math + h update (each thread owns its (m, c) cells) ----
    {
      int c = wid * 16 + ncol;
      float bcr = sb[64 + c], bcz = sb[128 + c];
      float bxn = sb[192 + c], bhn = sb[256 + c];
#pragma unroll
      for (int v = 0; v < 8; ++v) {
        int m = v + mrow0;
        float r = sigmoidf_(gx[0][v] + gh[0][v] + bcr);
        float z = sigmoidf_(gx[1][v] + gh[1][v] + bcz);
        float n = tanhf(gx[2][v] + bxn + r * (gh[2][v] + bhn));
        float hold = hf[m * 64 + c];
        float hnew = (1.0f - z) * n + z * hold;
        hf[m * 64 + c]   = hnew;
        Acat[m * 96 + c] = (bf16)hnew;   // bf16 mirror for next-step matmuls
      }
    }
    // next iteration's first __syncthreads() makes the h update visible
  }
}

extern "C" void kernel_launch(void* const* d_in, const int* in_sizes, int n_in,
                              void* d_out, int out_size, void* d_ws, size_t ws_size,
                              hipStream_t stream) {
  const float* problems = (const float*)d_in[0];
  const float* answers  = (const float*)d_in[1];
  const float* corrects = (const float*)d_in[2];
  const float* enc_w1   = (const float*)d_in[3];
  const float* enc_b1   = (const float*)d_in[4];
  const float* enc_w2   = (const float*)d_in[5];
  const float* enc_b2   = (const float*)d_in[6];
  const float* proj_w   = (const float*)d_in[7];
  const float* proj_b   = (const float*)d_in[8];
  const float* gru_wih  = (const float*)d_in[9];
  const float* gru_bih  = (const float*)d_in[10];
  const float* gru_whh  = (const float*)d_in[11];
  const float* gru_bhh  = (const float*)d_in[12];
  const float* pred_w1  = (const float*)d_in[13];
  const float* pred_b1  = (const float*)d_in[14];
  const float* pred_w2  = (const float*)d_in[15];
  const float* pred_b2  = (const float*)d_in[16];
  const float* ans_w    = (const float*)d_in[17];
  const float* ans_b    = (const float*)d_in[18];
  const float* cor_w    = (const float*)d_in[19];
  const float* cor_b    = (const float*)d_in[20];

  bf16*  frags  = (bf16*)d_ws;
  float* biases = (float*)((char*)d_ws + FRAG_BYTES);

  // Allow 80 KB of dynamic LDS for the weight-fragment cache (idempotent).
  (void)hipFuncSetAttribute((const void*)fused_scan,
                            hipFuncAttributeMaxDynamicSharedMemorySize,
                            FRAG_BYTES);

  prep_weights<<<1, 256, 0, stream>>>(enc_w1, enc_w2, proj_w, proj_b,
                                      gru_wih, gru_bih, gru_whh, gru_bhh,
                                      pred_w1, pred_w2, ans_w, cor_w,
                                      enc_b1, enc_b2, pred_b1, pred_b2,
                                      ans_b, cor_b, frags, biases);

  float* out_ans = (float*)d_out;
  float* out_cor = out_ans + (long)BB * TT * NB;
  fused_scan<<<BB / 16, 128, FRAG_BYTES, stream>>>(problems, answers, corrects,
                                                   frags, biases, out_ans, out_cor);
}
